// SPConv_32298154066078
// MI455X (gfx1250) — compile-verified, matrix operand order checked by reading, not run on hardware
//
#include <hip/hip_runtime.h>

typedef __attribute__((ext_vector_type(16))) __bf16 v16bf;
typedef __attribute__((ext_vector_type(8)))  float  v8f;

union Frag {
  v16bf bf;
  uint4 u4[2];
};

// Native hardware f32 -> bf16 convert (gfx1250 has BF16 VALU support; clang
// lowers fptrunc to the packed convert instruction).
__device__ __forceinline__ unsigned short f2bf(float f) {
  __bf16 h = (__bf16)f;
  return __builtin_bit_cast(unsigned short, h);
}

__device__ __forceinline__ v8f wmma_bf16(v16bf a, v16bf b, v8f c) {
  return __builtin_amdgcn_wmma_f32_16x16x32_bf16(false, a, false, b, (short)0, c,
                                                 false, false);
}

// Load a 16x32 bf16 A fragment from an LDS tile row-major region.
// lane l holds row (l%16); its 16 halves are two contiguous 8-half runs:
//   [colBase + (l/16)*8 .. +8) and [colBase + 16 + (l/16)*8 .. +8)
__device__ __forceinline__ v16bf load_a_frag(const unsigned short* rowPtr,
                                             int colBase, int laneHi) {
  Frag f;
  f.u4[0] = *(const uint4*)(rowPtr + colBase + laneHi * 8);
  f.u4[1] = *(const uint4*)(rowPtr + colBase + 16 + laneHi * 8);
  return f.bf;
}

// Pre-swizzled B fragments in global memory: ((nt*kTiles + ks)*32 + lane)*16 halves
__device__ __forceinline__ v16bf load_b_frag(const unsigned short* fw, int kTiles,
                                             int nt, int ks, int lane) {
  const uint4* p = (const uint4*)(fw + (((nt * kTiles + ks) * 32) + lane) * 16);
  Frag f;
  f.u4[0] = p[0];
  f.u4[1] = p[1];
  return f.bf;
}

// ---------------------------------------------------------------------------
// Prep: convert fp32 weight W[K][Ncols] (row-major, h @ W convention) into
// bf16 WMMA-B fragments laid out per (nt, ks, lane, idx).
__global__ void prep_wfrag_kernel(const float* __restrict__ W, int K, int Ncols,
                                  unsigned short* __restrict__ out) {
  int t = blockIdx.x * blockDim.x + threadIdx.x;
  if (t >= K * Ncols) return;
  int idx    = t & 15;
  int lane   = (t >> 4) & 31;
  int kTiles = K >> 5;
  int ks     = (t >> 9) % kTiles;
  int nt     = t / (512 * kTiles);
  int laneHi = lane >> 4;
  int n      = nt * 16 + (lane & 15);
  int k      = ks * 32 + idx + (idx < 8 ? 0 : 8) + laneHi * 8;
  out[t] = f2bf(W[k * Ncols + n]);
}

__global__ void zero_kernel(float* __restrict__ p, long n) {
  long i = (long)blockIdx.x * blockDim.x + threadIdx.x;
  if (i < n) p[i] = 0.0f;
}

// ---------------------------------------------------------------------------
// Edge kernel: per 16-edge tile per wave.
//   e  = relu(relu(delta @ We1 + be1) @ We2 + be2)     (K=3 in VALU, K=64 WMMA)
//   m  = relu([z_s, z_d, e] @ Wm1 + bm1) @ Wm2 + bm2   (K=192, K=64 WMMA)
//   atomicAdd sums[dst] += m ; cnt[dst] += 1
__global__ void __launch_bounds__(256)
edge_kernel(const float* __restrict__ z, const float* __restrict__ cent,
            const int* __restrict__ ei, const float* __restrict__ We1,
            const float* __restrict__ be1, const float* __restrict__ be2,
            const float* __restrict__ bm1, const float* __restrict__ bm2,
            const unsigned short* __restrict__ fWe2,
            const unsigned short* __restrict__ fWm1,
            const unsigned short* __restrict__ fWm2,
            float* __restrict__ sums, float* __restrict__ cnt, int N, int E) {
  constexpr int WAVES = 8;
  constexpr int HSTR  = 200;  // 192 cols + pad, keeps 16B alignment per row
  __shared__ alignas(16) unsigned short lds[WAVES][16][HSTR];

  const int wave   = threadIdx.x >> 5;
  const int lane   = threadIdx.x & 31;
  const int laneHi = lane >> 4;
  const int nlane  = lane & 15;
  const int ebase  = (blockIdx.x * WAVES + wave) * 16;
  if (ebase >= E) return;  // wave-uniform

  const int* __restrict__ src = ei;
  const int* __restrict__ dst = ei + E;

  // ---- gather z[src] -> cols [0,64), z[dst] -> cols [64,128) as bf16 ----
  {
    int r    = lane >> 1;   // row 0..15
    int half = lane & 1;    // 32-col half
    int e    = ebase + r;
    if (e >= E) e = E - 1;
    int si = src[e], di = dst[e];
    const float4* zs = (const float4*)(z + (long)si * 64 + half * 32);
    const float4* zd = (const float4*)(z + (long)di * 64 + half * 32);
    unsigned short* row = lds[wave][r];
#pragma unroll
    for (int j = 0; j < 8; ++j) {
      float4 a = zs[j];
      float4 b = zd[j];
      int c = half * 32 + j * 4;
      row[c + 0] = f2bf(a.x); row[c + 1] = f2bf(a.y);
      row[c + 2] = f2bf(a.z); row[c + 3] = f2bf(a.w);
      row[64 + c + 0] = f2bf(b.x); row[64 + c + 1] = f2bf(b.y);
      row[64 + c + 2] = f2bf(b.z); row[64 + c + 3] = f2bf(b.w);
    }
  }

  // ---- edge geometry layer (K=3): e1 = relu(delta @ We1 + be1) -> cols [128,192) ----
  {
    int e = ebase + nlane;
    if (e >= E) e = E - 1;
    int si = src[e], di = dst[e];
    float d0 = cent[(long)di * 3 + 0] - cent[(long)si * 3 + 0];
    float d1 = cent[(long)di * 3 + 1] - cent[(long)si * 3 + 1];
    float d2 = cent[(long)di * 3 + 2] - cent[(long)si * 3 + 2];
    unsigned short* row = lds[wave][nlane];
#pragma unroll
    for (int c = 0; c < 32; ++c) {
      int col   = laneHi * 32 + c;
      float acc = be1[col];
      acc = fmaf(d0, We1[col], acc);
      acc = fmaf(d1, We1[64 + col], acc);
      acc = fmaf(d2, We1[128 + col], acc);
      row[128 + col] = f2bf(fmaxf(acc, 0.0f));
    }
  }

  const unsigned short* arow = lds[wave][nlane];

  // ---- e2 = relu(e1 @ We2 + be2), overwrite cols [128,192) ----
  {
    v16bf a0 = load_a_frag(arow, 128, laneHi);
    v16bf a1 = load_a_frag(arow, 160, laneHi);
#pragma unroll
    for (int nt = 0; nt < 4; ++nt) {
      v8f acc = {};
      acc = wmma_bf16(a0, load_b_frag(fWe2, 2, nt, 0, lane), acc);
      acc = wmma_bf16(a1, load_b_frag(fWe2, 2, nt, 1, lane), acc);
      int col   = nt * 16 + nlane;
      float bias = be2[col];
#pragma unroll
      for (int v = 0; v < 8; ++v) {
        float x = fmaxf(acc[v] + bias, 0.0f);
        lds[wave][v + 8 * laneHi][128 + col] = f2bf(x);
      }
    }
  }

  // ---- m1 = relu(h @ Wm1 + bm1), h = cols [0,192); write m1 -> cols [0,64) ----
  {
    v16bf ha[6];
#pragma unroll
    for (int ks = 0; ks < 6; ++ks) ha[ks] = load_a_frag(arow, ks * 32, laneHi);
#pragma unroll
    for (int nt = 0; nt < 4; ++nt) {
      v8f acc = {};
#pragma unroll
      for (int ks = 0; ks < 6; ++ks)
        acc = wmma_bf16(ha[ks], load_b_frag(fWm1, 6, nt, ks, lane), acc);
      int col   = nt * 16 + nlane;
      float bias = bm1[col];
#pragma unroll
      for (int v = 0; v < 8; ++v) {
        float x = fmaxf(acc[v] + bias, 0.0f);
        lds[wave][v + 8 * laneHi][col] = f2bf(x);
      }
    }
  }

  // ---- m = m1 @ Wm2 + bm2 ; scatter-add into sums/cnt ----
  {
    v16bf a0 = load_a_frag(arow, 0, laneHi);
    v16bf a1 = load_a_frag(arow, 32, laneHi);
    int drow[8];
    bool ok[8];
#pragma unroll
    for (int v = 0; v < 8; ++v) {
      int e   = ebase + v + 8 * laneHi;
      ok[v]   = (e < E);
      drow[v] = dst[ok[v] ? e : (E - 1)];
    }
#pragma unroll
    for (int nt = 0; nt < 4; ++nt) {
      v8f acc = {};
      acc = wmma_bf16(a0, load_b_frag(fWm2, 2, nt, 0, lane), acc);
      acc = wmma_bf16(a1, load_b_frag(fWm2, 2, nt, 1, lane), acc);
      int col   = nt * 16 + nlane;
      float bias = bm2[col];
#pragma unroll
      for (int v = 0; v < 8; ++v) {
        if (ok[v]) atomicAdd(&sums[(long)drow[v] * 64 + col], acc[v] + bias);
      }
    }
    if (laneHi == 0 && (ebase + nlane) < E) {
      atomicAdd(&cnt[dst[ebase + nlane]], 1.0f);
    }
  }
}

// ---------------------------------------------------------------------------
// Node kernel: out = relu([z, sums/max(cnt,1)] @ Wu1 + bu1), K=128 WMMA.
__global__ void __launch_bounds__(256)
node_kernel(const float* __restrict__ z, const float* __restrict__ sums,
            const float* __restrict__ cnt, const float* __restrict__ bu1,
            const unsigned short* __restrict__ fWu1, float* __restrict__ out,
            int N) {
  constexpr int WAVES = 8;
  constexpr int HSTR  = 136;  // 128 cols + pad
  __shared__ alignas(16) unsigned short lds[WAVES][16][HSTR];

  const int wave   = threadIdx.x >> 5;
  const int lane   = threadIdx.x & 31;
  const int laneHi = lane >> 4;
  const int nlane  = lane & 15;
  const int tbase  = (blockIdx.x * WAVES + wave) * 16;
  if (tbase >= N) return;  // wave-uniform

  // ---- stage [z | M] tile as bf16 ----
  {
    int r    = lane >> 1;
    int half = lane & 1;  // 0 -> z cols [0,64), 1 -> M cols [64,128)
    int row  = tbase + r;
    if (row >= N) row = N - 1;
    unsigned short* lrow = lds[wave][r];
    if (half == 0) {
      const float4* p = (const float4*)(z + (long)row * 64);
#pragma unroll
      for (int j = 0; j < 16; ++j) {
        float4 a = p[j];
        int c = j * 4;
        lrow[c + 0] = f2bf(a.x); lrow[c + 1] = f2bf(a.y);
        lrow[c + 2] = f2bf(a.z); lrow[c + 3] = f2bf(a.w);
      }
    } else {
      float inv = 1.0f / fmaxf(cnt[row], 1.0f);
      const float4* p = (const float4*)(sums + (long)row * 64);
#pragma unroll
      for (int j = 0; j < 16; ++j) {
        float4 a = p[j];
        int c = 64 + j * 4;
        lrow[c + 0] = f2bf(a.x * inv); lrow[c + 1] = f2bf(a.y * inv);
        lrow[c + 2] = f2bf(a.z * inv); lrow[c + 3] = f2bf(a.w * inv);
      }
    }
  }

  const unsigned short* arow = lds[wave][nlane];
  v16bf a[4];
#pragma unroll
  for (int ks = 0; ks < 4; ++ks) a[ks] = load_a_frag(arow, ks * 32, laneHi);

#pragma unroll
  for (int nt = 0; nt < 4; ++nt) {
    v8f acc = {};
#pragma unroll
    for (int ks = 0; ks < 4; ++ks)
      acc = wmma_bf16(a[ks], load_b_frag(fWu1, 4, nt, ks, lane), acc);
    int col   = nt * 16 + nlane;
    float bias = bu1[col];
#pragma unroll
    for (int v = 0; v < 8; ++v) {
      int row = tbase + v + 8 * laneHi;
      if (row < N) out[(long)row * 64 + col] = fmaxf(acc[v] + bias, 0.0f);
    }
  }
}

// ---------------------------------------------------------------------------
extern "C" void kernel_launch(void* const* d_in, const int* in_sizes, int n_in,
                              void* d_out, int out_size, void* d_ws, size_t ws_size,
                              hipStream_t stream) {
  const float* z    = (const float*)d_in[0];
  const float* cent = (const float*)d_in[1];
  const int*   ei   = (const int*)d_in[2];
  const float* We1  = (const float*)d_in[3];
  const float* be1  = (const float*)d_in[4];
  const float* We2  = (const float*)d_in[5];
  const float* be2  = (const float*)d_in[6];
  const float* Wm1  = (const float*)d_in[7];
  const float* bm1  = (const float*)d_in[8];
  const float* Wm2  = (const float*)d_in[9];
  const float* bm2  = (const float*)d_in[10];
  const float* Wu1  = (const float*)d_in[11];
  const float* bu1  = (const float*)d_in[12];

  const int N = in_sizes[0] / 64;
  const int E = in_sizes[2] / 2;

  // workspace layout
  float* sums = (float*)d_ws;            // N*64 f32
  float* cnt  = sums + (long)N * 64;     // N f32
  unsigned short* fWe2 = (unsigned short*)(cnt + N);  // 64*64 bf16
  unsigned short* fWm1 = fWe2 + 64 * 64;              // 192*64 bf16
  unsigned short* fWm2 = fWm1 + 192 * 64;             // 64*64 bf16
  unsigned short* fWu1 = fWm2 + 64 * 64;              // 128*64 bf16

  long nz = (long)N * 65;
  zero_kernel<<<(int)((nz + 255) / 256), 256, 0, stream>>>(sums, nz);

  prep_wfrag_kernel<<<(64 * 64 + 255) / 256, 256, 0, stream>>>(We2, 64, 64, fWe2);
  prep_wfrag_kernel<<<(192 * 64 + 255) / 256, 256, 0, stream>>>(Wm1, 192, 64, fWm1);
  prep_wfrag_kernel<<<(64 * 64 + 255) / 256, 256, 0, stream>>>(Wm2, 64, 64, fWm2);
  prep_wfrag_kernel<<<(128 * 64 + 255) / 256, 256, 0, stream>>>(Wu1, 128, 64, fWu1);

  int eblocks = (E + 127) / 128;  // 8 waves x 16 edges per block
  edge_kernel<<<eblocks, 256, 0, stream>>>(z, cent, ei, We1, be1, be2, bm1, bm2,
                                           fWe2, fWm1, fWm2, sums, cnt, N, E);

  int nblocks = (N + 127) / 128;  // 8 waves x 16 nodes per block
  node_kernel<<<nblocks, 256, 0, stream>>>(z, sums, cnt, bu1, fWu1, (float*)d_out, N);
}
